// CharTransformerLM_40492951666991
// MI455X (gfx1250) — compile-verified, matrix operand order checked by reading, not run on hardware
//
#include <hip/hip_runtime.h>

// ---------------------------------------------------------------------------
// CharTransformerLM forward + loss for MI455X (gfx1250, wave32, WMMA).
// - All GEMMs + both attention matmuls: v_wmma_f32_16x16x32_bf16
// - GEMMs: 128x64 block tile, double-buffered async LDS staging
//   (global_load_async_to_lds_b128 / ASYNCcnt), 8 WMMAs per wave per barrier.
// ---------------------------------------------------------------------------

typedef __bf16 bf16_t;
typedef __attribute__((ext_vector_type(16))) __bf16 v16bf;
typedef __attribute__((ext_vector_type(8)))  __bf16 v8bf;
typedef __attribute__((ext_vector_type(8)))  float  v8f;
typedef int v4i_vec __attribute__((vector_size(16)));   // matches builtin proto

constexpr int Bb = 8;
constexpr int T  = 1024;
constexpr int V  = 256;
constexpr int E  = 512;
constexpr int H  = 8;
constexpr int L  = 6;
constexpr int D  = 64;        // E/H
constexpr int F  = 2048;      // 4*E
constexpr int MROWS = Bb * T; // 8192

__device__ __forceinline__ bf16_t f2bf(float f) {
  unsigned u = __builtin_bit_cast(unsigned, f);
  unsigned r = u + 0x7FFFu + ((u >> 16) & 1u);   // round-to-nearest-even
  unsigned short hs = (unsigned short)(r >> 16);
  return __builtin_bit_cast(bf16_t, hs);
}

// 16x32 bf16 A/B fragment (wave32 CDNA5 layout): lane holds row (lane&15);
// K chunks {8*hi..8*hi+7} and {16+8*hi..23+8*hi}.
__device__ __forceinline__ v16bf load_frag(const bf16_t* base, int ld) {
  int lane = threadIdx.x & 31;
  int r = lane & 15, hi = lane >> 4;
  const bf16_t* p = base + (size_t)r * ld + 8 * hi;
  v8bf c0 = *reinterpret_cast<const v8bf*>(p);
  v8bf c1 = *reinterpret_cast<const v8bf*>(p + 16);
  return __builtin_shufflevector(c0, c1, 0,1,2,3,4,5,6,7,8,9,10,11,12,13,14,15);
}

__device__ __forceinline__ v8f wmma_bf16(v16bf a, v16bf b, v8f c) {
  return __builtin_amdgcn_wmma_f32_16x16x32_bf16(false, a, false, b,
                                                 (short)0, c, false, false);
}

// --- Async global->LDS copy (16B per lane), ASYNCcnt-tracked ---------------
__device__ __forceinline__ void async_copy16(const bf16_t* g, bf16_t* l) {
#if __has_builtin(__builtin_amdgcn_global_load_async_to_lds_b128)
  __builtin_amdgcn_global_load_async_to_lds_b128(
      (__attribute__((address_space(1))) v4i_vec*)g,
      (__attribute__((address_space(3))) v4i_vec*)l, 0, 0);
#else
  unsigned loff = (unsigned)(size_t)(__attribute__((address_space(3))) bf16_t*)l;
  asm volatile("global_load_async_to_lds_b128 %0, %1, off"
               :: "v"(loff), "v"(g) : "memory");
#endif
}

__device__ __forceinline__ void wait_async0() {
#if __has_builtin(__builtin_amdgcn_s_wait_asynccnt)
  __builtin_amdgcn_s_wait_asynccnt(0);
#else
  asm volatile("s_wait_asynccnt 0" ::: "memory");
#endif
}

// ---------------------------------------------------------------------------
// Elementwise kernels
// ---------------------------------------------------------------------------

__global__ void embed_kernel(const int* __restrict__ idx,
                             const float* __restrict__ tok,
                             const float* __restrict__ pos,
                             float* __restrict__ x) {
  int i  = blockIdx.x * 256 + threadIdx.x;  // over MROWS*E
  int e  = i & (E - 1);
  int bt = i >> 9;                          // /E (E=512)
  int t  = bt & (T - 1);
  x[i] = tok[(size_t)idx[bt] * E + e] + pos[(size_t)t * E + e];
}

// W[l][k][n] (f32) -> Wt[l][n][k] (bf16)
__global__ void cvt_t_kernel(const float* __restrict__ W, bf16_t* __restrict__ Wt,
                             int K, int N, int total) {
  int i = blockIdx.x * 256 + threadIdx.x;
  if (i >= total) return;
  int kn = K * N;
  int l = i / kn, rem = i - l * kn;
  int n = rem / K, kk = rem - n * K;
  Wt[i] = f2bf(W[(size_t)l * kn + (size_t)kk * N + n]);
}

// LayerNorm: one wave per row of E=512; bf16 out.
__global__ void ln_kernel(const float* __restrict__ x,
                          const float* __restrict__ g,
                          const float* __restrict__ bta,
                          bf16_t* __restrict__ out) {
  int w = threadIdx.x >> 5, lane = threadIdx.x & 31;
  int row = blockIdx.x * 8 + w;
  const float* xr = x + (size_t)row * E;
  float v[16];
  float s = 0.f, ss = 0.f;
#pragma unroll
  for (int i = 0; i < 16; ++i) {
    v[i] = xr[lane + i * 32];
    s += v[i]; ss += v[i] * v[i];
  }
#pragma unroll
  for (int off = 1; off < 32; off <<= 1) {
    s  += __shfl_xor(s,  off, 32);
    ss += __shfl_xor(ss, off, 32);
  }
  float mu  = s * (1.f / E);
  float var = ss * (1.f / E) - mu * mu;
  float rs  = rsqrtf(var + 1e-5f);
  bf16_t* orow = out + (size_t)row * E;
#pragma unroll
  for (int i = 0; i < 16; ++i) {
    int c = lane + i * 32;
    orow[c] = f2bf((v[i] - mu) * rs * g[c] + bta[c]);
  }
}

// ---------------------------------------------------------------------------
// WMMA GEMM, 128x64 block tile, double-buffered async LDS staging (K-depth 64).
// C[M,N] = A[M,K] (bf16 rm) x Wt[N,K] (bf16).
// 8 waves; wave = 2 t-strips x 2 n-tiles = 4 C-frags; 8 WMMAs per stage.
// ---------------------------------------------------------------------------
enum { MODE_RES = 0, MODE_RELU = 1, MODE_QK = 2, MODE_VT = 3, MODE_OUT = 4 };

template <int NDIM, int KDIM, int MODE>
__global__ void gemm_kernel(const bf16_t* __restrict__ A,
                            const bf16_t* __restrict__ Wt,
                            const float* __restrict__ bias,
                            float* __restrict__ outF,
                            bf16_t* __restrict__ outB,
                            const float* __restrict__ resid,
                            float scale) {
  __shared__ __align__(16) bf16_t abuf[2][128 * 64];   // 2 x 16KB
  __shared__ __align__(16) bf16_t bbuf[2][64 * 64];    // 2 x  8KB
  const int tid = threadIdx.x;
  const int n0 = blockIdx.x * 64;
  const int m0 = blockIdx.y * 128;

  // stage a 128x64 A tile + 64x64 B tile (16B chunks; 6 async copies/thread)
  auto stage = [&](int kk, int bsel) {
#pragma unroll
    for (int i = 0; i < 4; ++i) {
      int chunk = tid + i * 256;             // 1024 chunks of A
      int row = chunk >> 3, off = (chunk & 7) * 8;
      async_copy16(A + (size_t)(m0 + row) * KDIM + kk + off,
                   &abuf[bsel][row * 64 + off]);
    }
#pragma unroll
    for (int i = 0; i < 2; ++i) {
      int chunk = tid + i * 256;             // 512 chunks of B
      int row = chunk >> 3, off = (chunk & 7) * 8;
      async_copy16(Wt + (size_t)(n0 + row) * KDIM + kk + off,
                   &bbuf[bsel][row * 64 + off]);
    }
  };

  stage(0, 0);
  wait_async0();
  __syncthreads();

  const int w = tid >> 5, lane = tid & 31;
  const int ts = (w & 3) * 32;    // t-strip pair base (rows ts, ts+16)
  const int np = (w >> 2) * 32;   // n-tile pair base (cols np, np+16)
  v8f c[2][2] = {};
  int buf = 0;
  for (int kk = 0; kk < KDIM; kk += 64) {
    if (kk + 64 < KDIM) stage(kk + 64, buf ^ 1);
#pragma unroll
    for (int ks = 0; ks < 64; ks += 32) {
      v16bf a0 = load_frag(&abuf[buf][ts * 64 + ks], 64);
      v16bf b0 = load_frag(&bbuf[buf][np * 64 + ks], 64);
      c[0][0] = wmma_bf16(a0, b0, c[0][0]);
      v16bf a1 = load_frag(&abuf[buf][(ts + 16) * 64 + ks], 64);
      c[1][0] = wmma_bf16(a1, b0, c[1][0]);
      v16bf b1 = load_frag(&bbuf[buf][(np + 16) * 64 + ks], 64);
      c[0][1] = wmma_bf16(a0, b1, c[0][1]);
      c[1][1] = wmma_bf16(a1, b1, c[1][1]);
    }
    wait_async0();                // next tile landed (per-wave)
    __syncthreads();              // block-wide visibility + reads done
    buf ^= 1;
  }

  const int col = lane & 15, hi = lane >> 4;
#pragma unroll
  for (int si = 0; si < 2; ++si) {
#pragma unroll
    for (int ni = 0; ni < 2; ++ni) {
      v8f cc = c[si][ni];
      int n = n0 + np + ni * 16 + col;
      float bs = 0.f;
      if constexpr (MODE == MODE_RES || MODE == MODE_RELU || MODE == MODE_OUT)
        bs = bias[n];
#pragma unroll
      for (int r = 0; r < 8; ++r) {
        int m = m0 + ts + si * 16 + r + 8 * hi;
        float vv = cc[r] + bs;
        if constexpr (MODE == MODE_RES) {
          size_t o = (size_t)m * NDIM + n;
          outF[o] = resid[o] + vv;
        } else if constexpr (MODE == MODE_RELU) {
          outB[(size_t)m * NDIM + n] = f2bf(fmaxf(vv, 0.f));
        } else if constexpr (MODE == MODE_QK) {
          int b = m >> 10, t = m & (T - 1);
          int hh = n >> 6, d = n & (D - 1);
          outB[((size_t)(b * H + hh) * T + t) * D + d] = f2bf(vv * scale);
        } else if constexpr (MODE == MODE_VT) {
          int b = m >> 10, t = m & (T - 1);
          int hh = n >> 6, d = n & (D - 1);
          outB[((size_t)(b * H + hh) * D + d) * T + t] = f2bf(vv);
        } else {  // MODE_OUT
          outF[(size_t)m * NDIM + n] = vv;
        }
      }
    }
  }
}

// ---------------------------------------------------------------------------
// Flash attention: grid (T/64, B*H), 128 threads (4 waves).
// Wave owns 16 query rows; online softmax; P converted C->A layout via a
// 1KB per-wave LDS bounce. Loads interleaved with WMMAs to curb VGPR spills.
// q pre-scaled by 1/sqrt(D).
// ---------------------------------------------------------------------------
__global__ void attention_kernel(const bf16_t* __restrict__ q,
                                 const bf16_t* __restrict__ k,
                                 const bf16_t* __restrict__ vt,
                                 bf16_t* __restrict__ attnb) {
  __shared__ __align__(16) bf16_t psh[4][16][32];
  int bh = blockIdx.y;                     // b*H + h
  int w = threadIdx.x >> 5, lane = threadIdx.x & 31;
  int col = lane & 15, hi = lane >> 4;
  int t0 = blockIdx.x * 64 + w * 16;

  const bf16_t* qp = q  + ((size_t)bh * T + t0) * D;
  const bf16_t* kp = k  + (size_t)bh * T * D;
  const bf16_t* vp = vt + (size_t)bh * D * T;

  v16bf qa0 = load_frag(qp,      D);       // d 0..31
  v16bf qa1 = load_frag(qp + 32, D);       // d 32..63

  float mrun[8], lrun[8];
#pragma unroll
  for (int r = 0; r < 8; ++r) { mrun[r] = -3.0e38f; lrun[r] = 0.f; }
  v8f o[4] = {};

  int smax = t0 + 16;                      // need s <= t0+15
  for (int s0 = 0; s0 < smax; s0 += 32) {
    __builtin_prefetch(kp + (size_t)(s0 + 32) * D, 0, 0);
    // S = Q K^T, interleaved load/WMMA
    v8f sA = {}, sB = {};
    {
      v16bf kf0 = load_frag(kp + (size_t)s0 * D, D);
      sA = wmma_bf16(qa0, kf0, sA);
      v16bf kf1 = load_frag(kp + (size_t)s0 * D + 32, D);
      sA = wmma_bf16(qa1, kf1, sA);
      v16bf kf2 = load_frag(kp + (size_t)(s0 + 16) * D, D);
      sB = wmma_bf16(qa0, kf2, sB);
      v16bf kf3 = load_frag(kp + (size_t)(s0 + 16) * D + 32, D);
      sB = wmma_bf16(qa1, kf3, sB);
    }
    if (s0 + 31 > t0) {                    // causal mask on diagonal tiles
#pragma unroll
      for (int r = 0; r < 8; ++r) {
        int t = t0 + r + 8 * hi;
        sA[r] = (s0 + col      <= t) ? sA[r] : -3.0e38f;
        sB[r] = (s0 + 16 + col <= t) ? sB[r] : -3.0e38f;
      }
    }
    // Online softmax update (row stats replicated across 16-lane halves)
    float cs[8];
#pragma unroll
    for (int r = 0; r < 8; ++r) {
      float mx = fmaxf(sA[r], sB[r]);
#pragma unroll
      for (int off = 1; off < 16; off <<= 1) mx = fmaxf(mx, __shfl_xor(mx, off, 32));
      float nm = fmaxf(mrun[r], mx);
      float pA = __expf(sA[r] - nm);
      float pB = __expf(sB[r] - nm);
      sA[r] = pA; sB[r] = pB;
      float sum = pA + pB;
#pragma unroll
      for (int off = 1; off < 16; off <<= 1) sum += __shfl_xor(sum, off, 32);
      cs[r]   = __expf(mrun[r] - nm);
      lrun[r] = lrun[r] * cs[r] + sum;
      mrun[r] = nm;
    }
#pragma unroll
    for (int r = 0; r < 8; ++r) {
      o[0][r] *= cs[r]; o[1][r] *= cs[r]; o[2][r] *= cs[r]; o[3][r] *= cs[r];
    }
    // C-layout P -> A-layout via per-wave LDS bounce (wave-local ordering)
#pragma unroll
    for (int r = 0; r < 8; ++r) {
      psh[w][r + 8 * hi][col]      = f2bf(sA[r]);
      psh[w][r + 8 * hi][col + 16] = f2bf(sB[r]);
    }
    asm volatile("s_wait_dscnt 0" ::: "memory");
    v16bf pa = load_frag(&psh[w][0][0], 32);
    // O += P V, interleaved load/WMMA (4 d-tiles)
#pragma unroll
    for (int dt = 0; dt < 4; ++dt) {
      v16bf vf = load_frag(vp + (size_t)(dt * 16) * T + s0, T);
      o[dt] = wmma_bf16(pa, vf, o[dt]);
    }
  }
  // Normalize and scatter to [B,T,E] (e = h*64 + dtile*16 + col)
  int b = bh >> 3, h = bh & (H - 1);
#pragma unroll
  for (int r = 0; r < 8; ++r) {
    float inv = 1.f / lrun[r];
    int t = t0 + r + 8 * hi;
    size_t base = ((size_t)(b * T + t)) * E + (size_t)h * D + col;
    attnb[base + 0 ] = f2bf(o[0][r] * inv);
    attnb[base + 16] = f2bf(o[1][r] * inv);
    attnb[base + 32] = f2bf(o[2][r] * inv);
    attnb[base + 48] = f2bf(o[3][r] * inv);
  }
}

// ---------------------------------------------------------------------------
// Loss: deterministic two-stage reduction of -mean(logp[target]).
// ---------------------------------------------------------------------------
__global__ void loss_partial_kernel(const float* __restrict__ logits,
                                    const int* __restrict__ tgt,
                                    float* __restrict__ partial) {
  __shared__ float wsum[8];
  int w = threadIdx.x >> 5, lane = threadIdx.x & 31;
  float acc = 0.f;
  for (int r = blockIdx.x * 8 + w; r < MROWS; r += gridDim.x * 8) {
    const float* row = logits + (size_t)r * V;
    float vals[8];
    float mx = -3.0e38f;
#pragma unroll
    for (int i = 0; i < 8; ++i) { vals[i] = row[lane + i * 32]; mx = fmaxf(mx, vals[i]); }
#pragma unroll
    for (int off = 1; off < 32; off <<= 1) mx = fmaxf(mx, __shfl_xor(mx, off, 32));
    float s = 0.f;
#pragma unroll
    for (int i = 0; i < 8; ++i) s += __expf(vals[i] - mx);
#pragma unroll
    for (int off = 1; off < 32; off <<= 1) s += __shfl_xor(s, off, 32);
    float logZ = mx + __logf(s);
    acc += logZ - row[tgt[r]];
  }
  if (lane == 0) wsum[w] = acc;
  __syncthreads();
  if (threadIdx.x == 0) {
    float s = 0.f;
    for (int i = 0; i < 8; ++i) s += wsum[i];
    partial[blockIdx.x] = s;
  }
}

__global__ void loss_final_kernel(const float* __restrict__ partial,
                                  float* __restrict__ loss) {
  if (threadIdx.x == 0 && blockIdx.x == 0) {
    float s = 0.f;
    for (int i = 0; i < 64; ++i) s += partial[i];
    *loss = s / (float)MROWS;
  }
}

// ---------------------------------------------------------------------------
// Host driver
// ---------------------------------------------------------------------------
extern "C" void kernel_launch(void* const* d_in, const int* in_sizes, int n_in,
                              void* d_out, int out_size, void* d_ws, size_t ws_size,
                              hipStream_t stream) {
  (void)in_sizes; (void)n_in; (void)out_size; (void)ws_size;

  const int*   idx     = (const int*)d_in[0];
  const int*   targets = (const int*)d_in[1];
  const float* tok_emb = (const float*)d_in[2];
  const float* pos_emb = (const float*)d_in[3];
  const float* ln1_w   = (const float*)d_in[4];
  const float* ln1_b   = (const float*)d_in[5];
  const float* ln2_w   = (const float*)d_in[6];
  const float* ln2_b   = (const float*)d_in[7];
  const float* wq      = (const float*)d_in[8];
  const float* wk      = (const float*)d_in[9];
  const float* wv      = (const float*)d_in[10];
  const float* wo      = (const float*)d_in[11];
  const float* wo_b    = (const float*)d_in[12];
  const float* w1      = (const float*)d_in[13];
  const float* w1_b    = (const float*)d_in[14];
  const float* w2      = (const float*)d_in[15];
  const float* w2_b    = (const float*)d_in[16];
  const float* lnf_w   = (const float*)d_in[17];
  const float* lnf_b   = (const float*)d_in[18];
  const float* head_w  = (const float*)d_in[19];
  const float* head_b  = (const float*)d_in[20];

  float* logits = (float*)d_out;                 // [MROWS, V]
  float* loss   = logits + (size_t)MROWS * V;

  char* p = (char*)d_ws;
  auto carve = [&](size_t bytes) {
    char* r = p; p += (bytes + 255) & ~(size_t)255; return (void*)r;
  };
  float*  x     = (float*) carve((size_t)MROWS * E * 4);
  bf16_t* hbuf  = (bf16_t*)carve((size_t)MROWS * E * 2);
  bf16_t* qb    = (bf16_t*)carve((size_t)MROWS * E * 2);
  bf16_t* kb    = (bf16_t*)carve((size_t)MROWS * E * 2);
  bf16_t* vtb   = (bf16_t*)carve((size_t)MROWS * E * 2);
  bf16_t* attnb = (bf16_t*)carve((size_t)MROWS * E * 2);
  bf16_t* mlp   = (bf16_t*)carve((size_t)MROWS * F * 2);
  bf16_t* wqt   = (bf16_t*)carve((size_t)L * E * E * 2);
  bf16_t* wkt   = (bf16_t*)carve((size_t)L * E * E * 2);
  bf16_t* wvt   = (bf16_t*)carve((size_t)L * E * E * 2);
  bf16_t* wot   = (bf16_t*)carve((size_t)L * E * E * 2);
  bf16_t* w1t   = (bf16_t*)carve((size_t)L * E * F * 2);
  bf16_t* w2t   = (bf16_t*)carve((size_t)L * F * E * 2);
  bf16_t* hwt   = (bf16_t*)carve((size_t)E * V * 2);
  float*  partial = (float*)carve(64 * 4);

  {
    int tEE = L * E * E, tEF = L * E * F, tEV = E * V;
    cvt_t_kernel<<<(tEE + 255) / 256, 256, 0, stream>>>(wq, wqt, E, E, tEE);
    cvt_t_kernel<<<(tEE + 255) / 256, 256, 0, stream>>>(wk, wkt, E, E, tEE);
    cvt_t_kernel<<<(tEE + 255) / 256, 256, 0, stream>>>(wv, wvt, E, E, tEE);
    cvt_t_kernel<<<(tEE + 255) / 256, 256, 0, stream>>>(wo, wot, E, E, tEE);
    cvt_t_kernel<<<(tEF + 255) / 256, 256, 0, stream>>>(w1, w1t, E, F, tEF);
    cvt_t_kernel<<<(tEF + 255) / 256, 256, 0, stream>>>(w2, w2t, F, E, tEF);
    cvt_t_kernel<<<(tEV + 255) / 256, 256, 0, stream>>>(head_w, hwt, E, V, tEV);
  }

  embed_kernel<<<(MROWS * E) / 256, 256, 0, stream>>>(idx, tok_emb, pos_emb, x);

  const dim3 gemmBlk(256);
  const dim3 gEE(E / 64, MROWS / 128);
  const dim3 gEF(F / 64, MROWS / 128);
  const dim3 gEV(V / 64, MROWS / 128);
  const float qscale = 0.125f;             // 1/sqrt(D)

  for (int l = 0; l < L; ++l) {
    ln_kernel<<<MROWS / 8, 256, 0, stream>>>(x, ln1_w + l * E, ln1_b + l * E, hbuf);
    gemm_kernel<E, E, MODE_QK><<<gEE, gemmBlk, 0, stream>>>(
        hbuf, wqt + (size_t)l * E * E, nullptr, nullptr, qb, nullptr, qscale);
    gemm_kernel<E, E, MODE_QK><<<gEE, gemmBlk, 0, stream>>>(
        hbuf, wkt + (size_t)l * E * E, nullptr, nullptr, kb, nullptr, 1.f);
    gemm_kernel<E, E, MODE_VT><<<gEE, gemmBlk, 0, stream>>>(
        hbuf, wvt + (size_t)l * E * E, nullptr, nullptr, vtb, nullptr, 1.f);
    attention_kernel<<<dim3(T / 64, Bb * H), 128, 0, stream>>>(qb, kb, vtb, attnb);
    gemm_kernel<E, E, MODE_RES><<<gEE, gemmBlk, 0, stream>>>(
        attnb, wot + (size_t)l * E * E, wo_b + l * E, x, nullptr, x, 1.f);
    ln_kernel<<<MROWS / 8, 256, 0, stream>>>(x, ln2_w + l * E, ln2_b + l * E, hbuf);
    gemm_kernel<F, E, MODE_RELU><<<gEF, gemmBlk, 0, stream>>>(
        hbuf, w1t + (size_t)l * E * F, w1_b + l * F, nullptr, mlp, nullptr, 1.f);
    gemm_kernel<E, F, MODE_RES><<<gEE, gemmBlk, 0, stream>>>(
        mlp, w2t + (size_t)l * F * E, w2_b + l * E, x, nullptr, x, 1.f);
  }

  ln_kernel<<<MROWS / 8, 256, 0, stream>>>(x, lnf_w, lnf_b, hbuf);
  gemm_kernel<V, E, MODE_OUT><<<gEV, gemmBlk, 0, stream>>>(
      hbuf, hwt, head_b, logits, nullptr, nullptr, 1.f);

  loss_partial_kernel<<<64, 256, 0, stream>>>(logits, targets, partial);
  loss_final_kernel<<<1, 64, 0, stream>>>(partial, loss);
}
